// MultiHeadAttention_Rope3D_bias_9732395892966
// MI455X (gfx1250) — compile-verified, matrix operand order checked by reading, not run on hardware
//
#include <hip/hip_runtime.h>
#include <cstdint>

// ---------------- CDNA5 (gfx1250) attention pipeline, wave32 + WMMA + TDM ----------------

#define USE_ASYNC_LDS 1
#define USE_TDM       1

typedef __attribute__((ext_vector_type(16))) __bf16       v16bf;
typedef __attribute__((ext_vector_type(8)))  __bf16       v8bf;
typedef __attribute__((ext_vector_type(8)))  float        v8f;
typedef __attribute__((ext_vector_type(4)))  unsigned int v4u;
typedef __attribute__((ext_vector_type(8)))  unsigned int v8u;

#define B_    4
#define S_    1024
#define D_    512
#define H_    8
#define DK_   64
#define NT_   402
#define FEAT_ 6
#define NEGV  (-1e9f)

// ---- WMMA wrapper: D = A(16x32 bf16) * B + C(16x16 f32) ----
__device__ __forceinline__ v8f wmma_bf16(v16bf a, v16bf b, v8f c) {
  return __builtin_amdgcn_wmma_f32_16x16x32_bf16(
      /*neg_a=*/false, a, /*neg_b=*/false, b,
      /*c_mod=*/(short)0, c, /*reuse_a=*/false, /*reuse_b=*/false);
}

// Per-lane fragment from a row-major [16][32] bf16 tile (ISA 7.12.2 layout):
//   lanes 0-15 : K = 0..7 and 16..23 ; lanes 16-31: K = 8..15 and 24..31
__device__ __forceinline__ v16bf ldfrag(const __bf16* row, int lane) {
  const int e0 = (lane & 16) ? 8 : 0;
  v8bf lo = *(const v8bf*)(row + e0);
  v8bf hi = *(const v8bf*)(row + 16 + e0);
  v16bf r;
#pragma unroll
  for (int i = 0; i < 8; ++i) { r[i] = lo[i]; r[8 + i] = hi[i]; }
  return r;
}

// ---- async global->LDS (ASYNCcnt) ----
__device__ __forceinline__ void async_ld_b32(void* lds_dst, const void* gsrc) {
#if USE_ASYNC_LDS
  unsigned int       l = (unsigned int)(unsigned long long)(size_t)lds_dst;
  unsigned long long g = (unsigned long long)(size_t)gsrc;
  asm volatile("global_load_async_to_lds_b32 %0, %1, off"
               :: "v"(l), "v"(g) : "memory");
#else
  *(unsigned int*)lds_dst = *(const unsigned int*)gsrc;
#endif
}
__device__ __forceinline__ void wait_async0() {
#if USE_ASYNC_LDS
  asm volatile("s_wait_asynccnt 0x0" ::: "memory");
#endif
}

// ---- TDM wait (TENSORcnt) ----
__device__ __forceinline__ void wait_tensor0() {
#if defined(__has_builtin) && __has_builtin(__builtin_amdgcn_s_wait_tensorcnt)
  __builtin_amdgcn_s_wait_tensorcnt(0);
#else
  asm volatile("s_wait_tensorcnt 0x0" ::: "memory");
#endif
}

// 2D TDM load: tile_d1 rows x tile_d0 elems (fp32), global row stride
// `stride0` elems, packed contiguously at `lds_dst`.  Issue from ONE wave.
__device__ __forceinline__ void tdm_load_2d_f32(void* lds_dst, const float* gsrc,
                                                unsigned tile_d0, unsigned tile_d1,
                                                unsigned stride0) {
  unsigned long long ga = (unsigned long long)(size_t)gsrc;
  v4u g0;
  g0[0] = 1u;                                        // count=1 user descriptor
  g0[1] = (unsigned int)(unsigned long long)(size_t)lds_dst;  // lds_addr (bytes)
  g0[2] = (unsigned int)ga;                          // global_addr[31:0]
  g0[3] = (unsigned int)((ga >> 32) & 0x1FFFFFFull) | (2u << 30);  // [56:32] | type=2
  v8u g1;
  g1[0] = 2u << 16;                                  // data_size = 4B
  g1[1] = (tile_d0 & 0xFFFFu) << 16;                 // tensor_dim0[15:0]
  g1[2] = (tile_d0 >> 16) | ((tile_d1 & 0xFFFFu) << 16);  // dim0 hi | tensor_dim1 lo
  g1[3] = (tile_d1 >> 16) | (tile_d0 << 16);         // dim1 hi | tile_dim0
  g1[4] = tile_d1;                                   // tile_dim1 | tile_dim2=0
  g1[5] = stride0;                                   // tensor_dim0_stride[31:0]
  g1[6] = 0u;
  g1[7] = 0u;
  asm volatile("tensor_load_to_lds %0, %1" :: "s"(g0), "s"(g1) : "memory");
}

// =====================================================================
// Kernel 1: projections.  Y(4096x512) = X(f32->bf16) @ W(512x512).
// =====================================================================
__global__ void proj_kernel(const float* __restrict__ xq, const float* __restrict__ xk,
                            const float* __restrict__ xv, const float* __restrict__ wq,
                            const float* __restrict__ wk, const float* __restrict__ wv,
                            __bf16* __restrict__ qo, __bf16* __restrict__ ko,
                            __bf16* __restrict__ vTo) {
  __shared__ __align__(16) __bf16 lA[128 * 32];
  __shared__ __align__(16) __bf16 lB[64 * 32];

  const int tid = threadIdx.x, lane = tid & 31, w = tid >> 5;
  const int m0 = blockIdx.x * 128, n0 = blockIdx.y * 64, mode = blockIdx.z;
  const float* X = (mode == 0) ? xq : (mode == 1) ? xk : xv;
  const float* W = (mode == 0) ? wq : (mode == 1) ? wk : wv;

  v8f z = {0.f, 0.f, 0.f, 0.f, 0.f, 0.f, 0.f, 0.f};
  v8f acc[4];
#pragma unroll
  for (int i = 0; i < 4; ++i) acc[i] = z;

  for (int kc = 0; kc < 512; kc += 32) {
    { // stage A: 128x32 f32 -> bf16
      const int r = tid >> 1, hc = (tid & 1) * 16;
      const float* src = X + (size_t)(m0 + r) * 512 + kc + hc;
      __bf16* dst = lA + r * 32 + hc;
      __builtin_prefetch(src + 32);
#pragma unroll
      for (int j = 0; j < 4; ++j) {
        float4 f = ((const float4*)src)[j];
        dst[j * 4 + 0] = (__bf16)f.x; dst[j * 4 + 1] = (__bf16)f.y;
        dst[j * 4 + 2] = (__bf16)f.z; dst[j * 4 + 3] = (__bf16)f.w;
      }
    }
    { // stage B transposed: lB[n][k] = W[kc+k][n0+n]
      const int k = tid & 31, ng = tid >> 5;
      const float* src = W + (size_t)(kc + k) * 512 + n0 + ng * 8;
      float4 f0 = ((const float4*)src)[0];
      float4 f1 = ((const float4*)src)[1];
      lB[(ng * 8 + 0) * 32 + k] = (__bf16)f0.x; lB[(ng * 8 + 1) * 32 + k] = (__bf16)f0.y;
      lB[(ng * 8 + 2) * 32 + k] = (__bf16)f0.z; lB[(ng * 8 + 3) * 32 + k] = (__bf16)f0.w;
      lB[(ng * 8 + 4) * 32 + k] = (__bf16)f1.x; lB[(ng * 8 + 5) * 32 + k] = (__bf16)f1.y;
      lB[(ng * 8 + 6) * 32 + k] = (__bf16)f1.z; lB[(ng * 8 + 7) * 32 + k] = (__bf16)f1.w;
    }
    __syncthreads();
    v16bf af = ldfrag(lA + (w * 16 + (lane & 15)) * 32, lane);
#pragma unroll
    for (int ns = 0; ns < 4; ++ns) {
      v16bf bf = ldfrag(lB + (ns * 16 + (lane & 15)) * 32, lane);
      acc[ns] = wmma_bf16(af, bf, acc[ns]);
    }
    __syncthreads();
  }

  // --- epilogue: block-uniform batch/head, branch hoisted out of loops ---
  const int b  = m0 >> 10;            // 128 | 1024 -> uniform per block
  const int hh = n0 >> 6;             // 64-wide N tile -> uniform per block
  const int sbase = (m0 & 1023) + w * 16 + ((lane & 16) ? 8 : 0);
  const int nl0 = lane & 15;

  if (mode == 2) {                    // V^T: [b][h][dv][s]
    __bf16* dst = vTo + ((size_t)b * H_ + hh) * DK_ * S_;
#pragma unroll
    for (int ns = 0; ns < 4; ++ns)
#pragma unroll
      for (int r8 = 0; r8 < 8; ++r8)
        dst[(size_t)(ns * 16 + nl0) * S_ + (sbase + r8)] = (__bf16)acc[ns][r8];
  } else {                            // Q/K natural: [b][h][s][dk]
    __bf16* dst = ((mode == 0) ? qo : ko) + ((size_t)b * H_ + hh) * S_ * DK_;
#pragma unroll
    for (int ns = 0; ns < 4; ++ns)
#pragma unroll
      for (int r8 = 0; r8 < 8; ++r8)
        dst[(size_t)(sbase + r8) * DK_ + ns * 16 + nl0] = (__bf16)acc[ns][r8];
  }
}

// =====================================================================
// Kernel 2: scores = QK^T/8 + type_emb[itype] + relu(im @ w_feat), masked.
// Wave w == head h.  Interaction tile staged by the Tensor Data Mover
// (wave 0 issues a 2D D#: 16 rows x 96 fp32, stride S*6), itype tile via
// async-to-LDS loads.  Writes pre-softmax scores into the attn output.
// =====================================================================
__global__ void scores_kernel(const __bf16* __restrict__ qb, const __bf16* __restrict__ kb,
                              const float* __restrict__ im, const int* __restrict__ itype,
                              const unsigned char* __restrict__ aif,
                              const float* __restrict__ type_emb,
                              const float* __restrict__ w_feat,
                              float* __restrict__ attn) {
  __shared__ float ldsTE[NT_ * H_];                  // 12864 B
  __shared__ float ldsWF[FEAT_ * H_];
  __shared__ __align__(16) float ldsIM[256 * FEAT_]; // 6 KB, TDM dest
  __shared__ int ldsTY[256];
  __shared__ unsigned char ldsMK[256];

  const int tid = threadIdx.x, lane = tid & 31, h = tid >> 5;
  const int q0 = blockIdx.x * 16, b = blockIdx.y;

  for (int i = tid; i < NT_ * H_; i += 256) ldsTE[i] = type_emb[i];
  if (tid < FEAT_ * H_) ldsWF[tid] = w_feat[tid];
  __syncthreads();

  const __bf16* qrow = qb + (((size_t)b * H_ + h) * S_ + q0 + (lane & 15)) * DK_;
  v16bf a0 = ldfrag(qrow, lane);
  v16bf a1 = ldfrag(qrow + 32, lane);

  const int ql = tid >> 4, kl = tid & 15;

  for (int kt = 0; kt < 64; ++kt) {
    const int k0 = kt * 16;
    const size_t base = ((size_t)(b * S_ + q0 + ql)) * S_ + (k0 + kl);
    { // ---- stage tile: TDM for interaction matrix, async for itype ----
#if USE_TDM
      if (h == 0) {
        const float* tile0 = im + (((size_t)(b * S_ + q0)) * S_ + k0) * FEAT_;
        tdm_load_2d_f32(ldsIM, tile0, /*tile_d0=*/16 * FEAT_, /*tile_d1=*/16,
                        /*stride0=*/S_ * FEAT_);
      }
#else
      {
        const float* imp = im + base * FEAT_;
        float* dst = ldsIM + (ql * 16 + kl) * FEAT_;
#pragma unroll
        for (int f = 0; f < FEAT_; ++f) async_ld_b32(dst + f, imp + f);
      }
#endif
      async_ld_b32(&ldsTY[tid], &itype[base]);
      ldsMK[tid] = aif[base];
      wait_async0();
#if USE_TDM
      if (h == 0) wait_tensor0();
#endif
    }
    __syncthreads();

    const __bf16* krow = kb + (((size_t)b * H_ + h) * S_ + k0 + (lane & 15)) * DK_;
    v16bf b0 = ldfrag(krow, lane);
    v16bf b1 = ldfrag(krow + 32, lane);
    v8f acc = {0.f, 0.f, 0.f, 0.f, 0.f, 0.f, 0.f, 0.f};
    acc = wmma_bf16(a0, b0, acc);
    acc = wmma_bf16(a1, b1, acc);

#pragma unroll
    for (int r = 0; r < 8; ++r) {
      const int M = r + ((lane & 16) ? 8 : 0);
      const int N = lane & 15;
      const int li = M * 16 + N;
      float bias = ldsTE[ldsTY[li] * H_ + h];
      float s = 0.f;
#pragma unroll
      for (int f = 0; f < FEAT_; ++f) s += ldsIM[li * FEAT_ + f] * ldsWF[f * H_ + h];
      float val = ldsMK[li] ? NEGV : (acc[r] * 0.125f + bias + fmaxf(s, 0.f));
      attn[(((size_t)b * H_ + h) * S_ + q0 + M) * S_ + (k0 + N)] = val;
    }
    __syncthreads();
  }
}

// =====================================================================
// Kernel 3: in-place row softmax over 1024 elements.  Block 256 per row.
// =====================================================================
__global__ void softmax_kernel(float* __restrict__ attn) {
  __shared__ float red[256];
  const size_t row = blockIdx.x;
  const int tid = threadIdx.x;
  float* p = attn + row * (size_t)S_;
  float4 x = ((float4*)p)[tid];

  float m = fmaxf(fmaxf(x.x, x.y), fmaxf(x.z, x.w));
  red[tid] = m; __syncthreads();
  for (int s = 128; s > 0; s >>= 1) {
    if (tid < s) red[tid] = fmaxf(red[tid], red[tid + s]);
    __syncthreads();
  }
  m = red[0]; __syncthreads();

  x.x = __expf(x.x - m); x.y = __expf(x.y - m);
  x.z = __expf(x.z - m); x.w = __expf(x.w - m);
  red[tid] = x.x + x.y + x.z + x.w; __syncthreads();
  for (int s = 128; s > 0; s >>= 1) {
    if (tid < s) red[tid] += red[tid + s];
    __syncthreads();
  }
  const float inv = 1.f / red[0];
  x.x *= inv; x.y *= inv; x.z *= inv; x.w *= inv;
  ((float4*)p)[tid] = x;
}

// =====================================================================
// Kernel 4: context = attn (f32->bf16) @ V.
// =====================================================================
__global__ void context_kernel(const float* __restrict__ attn,
                               const __bf16* __restrict__ vT,
                               __bf16* __restrict__ cb) {
  __shared__ __align__(16) __bf16 lA[128 * 32];
  __shared__ __align__(16) __bf16 lB[64 * 32];

  const int tid = threadIdx.x, lane = tid & 31, w = tid >> 5;
  const int q0 = blockIdx.x * 128, bh = blockIdx.y;

  v8f z = {0.f, 0.f, 0.f, 0.f, 0.f, 0.f, 0.f, 0.f};
  v8f acc[4];
#pragma unroll
  for (int i = 0; i < 4; ++i) acc[i] = z;

  for (int kc = 0; kc < 1024; kc += 32) {
    {
      const int r = tid >> 1, hc = (tid & 1) * 16;
      const float* src = attn + ((size_t)bh * S_ + q0 + r) * S_ + kc + hc;
      __bf16* dst = lA + r * 32 + hc;
      __builtin_prefetch(src + 32);
#pragma unroll
      for (int j = 0; j < 4; ++j) {
        float4 f = ((const float4*)src)[j];
        dst[j * 4 + 0] = (__bf16)f.x; dst[j * 4 + 1] = (__bf16)f.y;
        dst[j * 4 + 2] = (__bf16)f.z; dst[j * 4 + 3] = (__bf16)f.w;
      }
    }
    {
      const int dv = tid >> 2, ch = (tid & 3) * 8;
      *(v8bf*)(lB + dv * 32 + ch) =
          *(const v8bf*)(vT + ((size_t)bh * DK_ + dv) * S_ + kc + ch);
    }
    __syncthreads();
    v16bf af = ldfrag(lA + (w * 16 + (lane & 15)) * 32, lane);
#pragma unroll
    for (int ns = 0; ns < 4; ++ns) {
      v16bf bf = ldfrag(lB + (ns * 16 + (lane & 15)) * 32, lane);
      acc[ns] = wmma_bf16(af, bf, acc[ns]);
    }
    __syncthreads();
  }

  const int b = bh >> 3, h = bh & 7;
  const int qbase = q0 + w * 16 + ((lane & 16) ? 8 : 0);
  __bf16* dst = cb + (size_t)b * S_ * D_ + h * DK_;
#pragma unroll
  for (int ns = 0; ns < 4; ++ns)
#pragma unroll
    for (int r8 = 0; r8 < 8; ++r8)
      dst[(size_t)(qbase + r8) * D_ + ns * 16 + (lane & 15)] = (__bf16)acc[ns][r8];
}

// =====================================================================
// Kernel 5: preln = context @ w_fc + residual.
// =====================================================================
__global__ void out_kernel(const __bf16* __restrict__ cb, const float* __restrict__ wfc,
                           const float* __restrict__ resid, float* __restrict__ preln) {
  __shared__ __align__(16) __bf16 lA[128 * 32];
  __shared__ __align__(16) __bf16 lB[64 * 32];

  const int tid = threadIdx.x, lane = tid & 31, w = tid >> 5;
  const int m0 = blockIdx.x * 128, n0 = blockIdx.y * 64;

  v8f z = {0.f, 0.f, 0.f, 0.f, 0.f, 0.f, 0.f, 0.f};
  v8f acc[4];
#pragma unroll
  for (int i = 0; i < 4; ++i) acc[i] = z;

  for (int kc = 0; kc < 512; kc += 32) {
    {
      const int r = tid >> 1, hc = (tid & 1) * 16;
      const __bf16* src = cb + (size_t)(m0 + r) * 512 + kc + hc;
      __builtin_prefetch(src + 32);
      *(v8bf*)(lA + r * 32 + hc + 0) = *(const v8bf*)(src + 0);
      *(v8bf*)(lA + r * 32 + hc + 8) = *(const v8bf*)(src + 8);
    }
    {
      const int k = tid & 31, ng = tid >> 5;
      const float* src = wfc + (size_t)(kc + k) * 512 + n0 + ng * 8;
      float4 f0 = ((const float4*)src)[0];
      float4 f1 = ((const float4*)src)[1];
      lB[(ng * 8 + 0) * 32 + k] = (__bf16)f0.x; lB[(ng * 8 + 1) * 32 + k] = (__bf16)f0.y;
      lB[(ng * 8 + 2) * 32 + k] = (__bf16)f0.z; lB[(ng * 8 + 3) * 32 + k] = (__bf16)f0.w;
      lB[(ng * 8 + 4) * 32 + k] = (__bf16)f1.x; lB[(ng * 8 + 5) * 32 + k] = (__bf16)f1.y;
      lB[(ng * 8 + 6) * 32 + k] = (__bf16)f1.z; lB[(ng * 8 + 7) * 32 + k] = (__bf16)f1.w;
    }
    __syncthreads();
    v16bf af = ldfrag(lA + (w * 16 + (lane & 15)) * 32, lane);
#pragma unroll
    for (int ns = 0; ns < 4; ++ns) {
      v16bf bf = ldfrag(lB + (ns * 16 + (lane & 15)) * 32, lane);
      acc[ns] = wmma_bf16(af, bf, acc[ns]);
    }
    __syncthreads();
  }

  const int mbase = m0 + w * 16 + ((lane & 16) ? 8 : 0);
#pragma unroll
  for (int ns = 0; ns < 4; ++ns)
#pragma unroll
    for (int r8 = 0; r8 < 8; ++r8) {
      const size_t idx = (size_t)(mbase + r8) * 512 + n0 + ns * 16 + (lane & 15);
      preln[idx] = acc[ns][r8] + resid[idx];
    }
}

// =====================================================================
// Kernel 6: LayerNorm over last dim (512).  Block 256 per row.
// =====================================================================
__global__ void ln_kernel(const float* __restrict__ preln, const float* __restrict__ g,
                          const float* __restrict__ bb, float* __restrict__ out) {
  __shared__ float rs[256];
  __shared__ float rq[256];
  const size_t row = blockIdx.x;
  const int tid = threadIdx.x;
  const float* p = preln + row * (size_t)D_;
  float x0 = p[tid], x1 = p[tid + 256];
  rs[tid] = x0 + x1;
  rq[tid] = x0 * x0 + x1 * x1;
  __syncthreads();
  for (int s = 128; s > 0; s >>= 1) {
    if (tid < s) { rs[tid] += rs[tid + s]; rq[tid] += rq[tid + s]; }
    __syncthreads();
  }
  const float mean = rs[0] * (1.f / (float)D_);
  const float var = rq[0] * (1.f / (float)D_) - mean * mean;
  const float rstd = rsqrtf(var + 1e-5f);
  out[row * D_ + tid]       = (x0 - mean) * rstd * g[tid] + bb[tid];
  out[row * D_ + tid + 256] = (x1 - mean) * rstd * g[tid + 256] + bb[tid + 256];
}

// =====================================================================
extern "C" void kernel_launch(void* const* d_in, const int* in_sizes, int n_in,
                              void* d_out, int out_size, void* d_ws, size_t ws_size,
                              hipStream_t stream) {
  (void)in_sizes; (void)n_in; (void)out_size; (void)ws_size;

  const float* input_Q  = (const float*)d_in[0];
  const float* input_K  = (const float*)d_in[1];
  const float* input_V  = (const float*)d_in[2];
  const unsigned char* attn_if = (const unsigned char*)d_in[4];
  const int*   itype    = (const int*)d_in[5];
  const float* im       = (const float*)d_in[6];
  const float* w_q      = (const float*)d_in[9];
  const float* w_k      = (const float*)d_in[10];
  const float* w_v      = (const float*)d_in[11];
  const float* w_fc     = (const float*)d_in[12];
  const float* ln_g     = (const float*)d_in[13];
  const float* ln_b     = (const float*)d_in[14];
  const float* type_emb = (const float*)d_in[15];
  const float* w_feat   = (const float*)d_in[16];

  float* out_f  = (float*)d_out;
  float* attn_f = out_f + (size_t)B_ * S_ * D_;   // second tuple element

  const size_t NEL = (size_t)B_ * H_ * S_ * DK_;  // 2,097,152
  __bf16* qb    = (__bf16*)d_ws;
  __bf16* kb    = qb + NEL;
  __bf16* vT    = kb + NEL;
  __bf16* cb    = vT + NEL;
  float*  preln = (float*)(cb + NEL);             // total ws use = 24 MB

  proj_kernel<<<dim3(32, 8, 3), 256, 0, stream>>>(
      input_Q, input_K, input_V, w_q, w_k, w_v, qb, kb, vT);

  scores_kernel<<<dim3(S_ / 16, B_), 256, 0, stream>>>(
      qb, kb, im, itype, attn_if, type_emb, w_feat, attn_f);

  softmax_kernel<<<dim3(B_ * H_ * S_), 256, 0, stream>>>(attn_f);

  context_kernel<<<dim3(S_ / 128, B_ * H_), 256, 0, stream>>>(attn_f, vT, cb);

  out_kernel<<<dim3(32, 8), 256, 0, stream>>>(cb, w_fc, input_Q, preln);

  ln_kernel<<<dim3(B_ * S_), 256, 0, stream>>>(preln, ln_g, ln_b, out_f);
}
